// ChannelAdaptedAttentionPerturbationModule_42520176230594
// MI455X (gfx1250) — compile-verified, matrix operand order checked by reading, not run on hardware
//
#include <hip/hip_runtime.h>
#include <hip/hip_bf16.h>

// ---------------------------------------------------------------------------
// CDNA5 (gfx1250, wave32) implementation of the channel-attention block.
// All matmuls go through v_wmma_f32_16x16x32_bf16 with 4x2 register blocking
// per wave (8 WMMAs per 12 global_load_b128 -> 2.7x fewer loads per WMMA than
// the naive 1-tile-per-wave form). Activations are kept in NHWC ([n][c]) bf16
// so A-rows and B-columns are contiguous; q/k are kept in NCHW for the C x C
// attention GEMM (K = 16384 contiguous).
// ---------------------------------------------------------------------------

typedef __attribute__((ext_vector_type(16))) __bf16 v16bf;
typedef __attribute__((ext_vector_type(8)))  float  v8f;

#define B_    4
#define C_    192
#define H_    128
#define W_    128
#define NSP   (H_ * W_)        // 16384
#define HID   510
#define HID2  1020
#define HPAD  512              // padded K for the fo GEMM (K % 32 == 0)
#define KKEEP 172              // max(1, int(C * 0.9))

static __device__ __forceinline__ float wave_sum(float v) {
    for (int o = 16; o > 0; o >>= 1) v += __shfl_xor(v, o, 32);
    return v;
}
static __device__ __forceinline__ float wave_max(float v) {
    for (int o = 16; o > 0; o >>= 1) v = fmaxf(v, __shfl_xor(v, o, 32));
    return v;
}
static __device__ __forceinline__ float wave_min(float v) {
    for (int o = 16; o > 0; o >>= 1) v = fminf(v, __shfl_xor(v, o, 32));
    return v;
}
static __device__ __forceinline__ int wave_imin(int v) {
    for (int o = 16; o > 0; o >>= 1) v = min(v, __shfl_xor(v, o, 32));
    return v;
}

// ---------------------------------------------------------------------------
// Generic WMMA GEMM:  D[m][n] = sum_k A[m][k] * X[n][k]  (+bias +scale +add)
// A: M x K rows (ld = K), optionally batched.  X: per-"n" rows of K (ld = K).
// One wave computes an MT x NT grid of 16x16 tiles; 8 waves per block tile N.
// Fragment layout per CDNA5 ISA 7.12.2 (16-bit A, wave32):
//   lanes 0-15: index = lane,  elems 0-7 -> K k0+0..7,  elems 8-15 -> k0+16..23
//   lanes 16-31: same index,   elems 0-7 -> K k0+8..15, elems 8-15 -> k0+24..31
// B uses the identical per-lane pattern with lane = N index (column-major B).
// ---------------------------------------------------------------------------
template <int MT, int NT>
__global__ __launch_bounds__(256) void gemm_wmma_kernel(
    const __bf16* __restrict__ A, long aBatch,
    const __bf16* __restrict__ X, long xBatch,
    const float* __restrict__ bias,
    const float* __restrict__ scale_ptr,
    const float* __restrict__ addsrc, long addBatch, int ldAdd,
    float* __restrict__ outF, __bf16* __restrict__ outH,
    long oBatch, int ldo, int outTrans,
    int M, int N, int K)
{
    const int wave = threadIdx.x >> 5;
    const int lane = threadIdx.x & 31;
    const int b    = blockIdx.z;
    const int m0   = blockIdx.x * (16 * MT);
    const int n0   = (blockIdx.y * 8 + wave) * (16 * NT);
    if (m0 >= M || n0 >= N) return;          // wave-uniform: EXEC stays all-ones

    const __bf16* Ab = A + (long)b * aBatch;
    const __bf16* Xb = X + (long)b * xBatch;
    const int hi8 = (lane >> 4) * 8;

    const __bf16* arow[MT];
    const __bf16* xrow[NT];
#pragma unroll
    for (int mi = 0; mi < MT; ++mi) {
        int r = m0 + mi * 16 + (lane & 15);
        if (r >= M) r = M - 1;
        arow[mi] = Ab + (long)r * K;
    }
#pragma unroll
    for (int ni = 0; ni < NT; ++ni) {
        int r = n0 + ni * 16 + (lane & 15);
        if (r >= N) r = N - 1;
        xrow[ni] = Xb + (long)r * K;
    }

    const v8f vzero = {};
    v8f acc[MT][NT];
#pragma unroll
    for (int mi = 0; mi < MT; ++mi)
#pragma unroll
        for (int ni = 0; ni < NT; ++ni) acc[mi][ni] = vzero;

    for (int k0 = 0; k0 < K; k0 += 32) {
        v16bf af[MT], bf[NT];
#pragma unroll
        for (int mi = 0; mi < MT; ++mi) {
            ((uint4*)&af[mi])[0] = *(const uint4*)(arow[mi] + k0 + hi8);
            ((uint4*)&af[mi])[1] = *(const uint4*)(arow[mi] + k0 + 16 + hi8);
        }
#pragma unroll
        for (int ni = 0; ni < NT; ++ni) {
            ((uint4*)&bf[ni])[0] = *(const uint4*)(xrow[ni] + k0 + hi8);
            ((uint4*)&bf[ni])[1] = *(const uint4*)(xrow[ni] + k0 + 16 + hi8);
            __builtin_prefetch(xrow[ni] + k0 + 128, 0, 0);   // global_prefetch_b8
        }
#pragma unroll
        for (int mi = 0; mi < MT; ++mi)
#pragma unroll
            for (int ni = 0; ni < NT; ++ni)
                acc[mi][ni] = __builtin_amdgcn_wmma_f32_16x16x32_bf16(
                    /*neg_a=*/false, af[mi], /*neg_b=*/false, bf[ni],
                    /*c_mod=*/(short)0, acc[mi][ni],
                    /*reuse_a=*/false, /*reuse_b=*/false);
    }

    const float scl = scale_ptr ? scale_ptr[0] : 1.0f;
#pragma unroll
    for (int mi = 0; mi < MT; ++mi) {
#pragma unroll
        for (int ni = 0; ni < NT; ++ni) {
            const int nn = n0 + ni * 16 + (lane & 15);
#pragma unroll
            for (int r = 0; r < 8; ++r) {
                const int m = m0 + mi * 16 + r + hi8;   // VGPR r -> M = r (+8 hi)
                if (m >= M || nn >= N) continue;
                float v = acc[mi][ni][r] * scl;
                if (bias)   v += bias[m];
                if (addsrc) v += addsrc[(long)b * addBatch + (long)nn * ldAdd + m];
                const long off = (long)b * oBatch +
                    (outTrans ? ((long)m * ldo + nn) : ((long)nn * ldo + m));
                if (outF) outF[off] = v;
                else      outH[off] = (__bf16)v;
            }
        }
    }
}

// NCHW f32 -> NHWC bf16
__global__ void nchw2nhwc_kernel(const float* __restrict__ src,
                                 __bf16* __restrict__ dst, long total)
{
    long idx = (long)blockIdx.x * blockDim.x + threadIdx.x;
    if (idx >= total) return;
    int  c = (int)(idx % C_);
    long t = idx / C_;
    long n = t % NSP;
    long b = t / NSP;
    dst[idx] = (__bf16)src[(b * C_ + c) * NSP + n];
}

__global__ void cast_f2b_kernel(const float* __restrict__ src,
                                __bf16* __restrict__ dst, long total)
{
    long idx = (long)blockIdx.x * blockDim.x + threadIdx.x;
    if (idx < total) dst[idx] = (__bf16)src[idx];
}

// fo_w (C x HID) f32 -> (C x HPAD) bf16, zero-padded
__global__ void pad_fow_kernel(const float* __restrict__ src,
                               __bf16* __restrict__ dst)
{
    long idx = (long)blockIdx.x * blockDim.x + threadIdx.x;
    if (idx >= (long)C_ * HPAD) return;
    int k = (int)(idx % HPAD);
    int c = (int)(idx / HPAD);
    dst[idx] = (k < HID) ? (__bf16)src[(long)c * HID + k] : (__bf16)0.f;
}

// depthwise 3x3 on NHWC bf16, +bias; outTrans: 0 -> NHWC bf16, 1 -> NCHW bf16
__global__ void dw3x3_kernel(const __bf16* __restrict__ in,
                             const float* __restrict__ wk,
                             const float* __restrict__ bias,
                             __bf16* __restrict__ out, int outTrans, long total)
{
    long idx = (long)blockIdx.x * blockDim.x + threadIdx.x;
    if (idx >= total) return;
    int  c = (int)(idx % C_);
    long t = idx / C_;
    int  x = (int)(t % W_);
    long t2 = t / W_;
    int  y = (int)(t2 % H_);
    int  b = (int)(t2 / H_);
    float s = bias[c];
#pragma unroll
    for (int ky = 0; ky < 3; ++ky) {
        int yy = y + ky - 1;
        if ((unsigned)yy >= (unsigned)H_) continue;
#pragma unroll
        for (int kx = 0; kx < 3; ++kx) {
            int xx = x + kx - 1;
            if ((unsigned)xx >= (unsigned)W_) continue;
            s += wk[c * 9 + ky * 3 + kx] *
                 (float)in[(((long)b * NSP) + (long)yy * W_ + xx) * C_ + c];
        }
    }
    long n = (long)y * W_ + x;
    if (outTrans) out[((long)b * C_ + c) * NSP + n] = (__bf16)s;
    else          out[idx] = (__bf16)s;
}

// sum of squares of one NCHW row (one block per (b,c) row)
__global__ __launch_bounds__(256) void rowsumsq_kernel(
    const __bf16* __restrict__ x, float* __restrict__ sums, int len)
{
    long base = (long)blockIdx.x * len;
    float s = 0.f;
    for (int i = threadIdx.x; i < len; i += 256) {
        float t = (float)x[base + i];
        s += t * t;
    }
    __shared__ float red[8];
    s = wave_sum(s);
    if ((threadIdx.x & 31) == 0) red[threadIdx.x >> 5] = s;
    __syncthreads();
    if (threadIdx.x == 0) {
        float t = 0.f;
        for (int i = 0; i < 8; ++i) t += red[i];
        sums[blockIdx.x] = t;
    }
}

__global__ void l2scale_kernel(__bf16* __restrict__ x,
                               const float* __restrict__ sums, long total)
{
    long idx = (long)blockIdx.x * blockDim.x + threadIdx.x;
    if (idx >= total) return;
    long row = idx / NSP;
    float scale = 1.f / fmaxf(sqrtf(sums[row]), 1e-12f);
    x[idx] = (__bf16)((float)x[idx] * scale);
}

// one wave per attention row: softmax then exact k-th-largest threshold mask
__global__ __launch_bounds__(256) void softmax_topk_kernel(
    const float* __restrict__ attn, __bf16* __restrict__ out, int rows)
{
    int row  = blockIdx.x * 8 + (threadIdx.x >> 5);
    int lane = threadIdx.x & 31;
    if (row >= rows) return;
    const float* rp = attn + (long)row * C_;
    float v[6], p[6], w[6];
#pragma unroll
    for (int j = 0; j < 6; ++j) v[j] = rp[lane + 32 * j];
    float m = v[0];
#pragma unroll
    for (int j = 1; j < 6; ++j) m = fmaxf(m, v[j]);
    m = wave_max(m);
    float s = 0.f;
#pragma unroll
    for (int j = 0; j < 6; ++j) { v[j] = __expf(v[j] - m); s += v[j]; }
    s = wave_sum(s);
    float inv = 1.f / s;
#pragma unroll
    for (int j = 0; j < 6; ++j) { p[j] = v[j] * inv; w[j] = p[j]; }

    // threshold = (C-KKEEP+1)-th smallest: remove (C-KKEEP) minima, take min
    const float BIG = 3.4e38f;
    float thr = 0.f;
    const int nrem = C_ - KKEEP;                 // 20
    for (int it = 0; it <= nrem; ++it) {
        float lm = w[0];
#pragma unroll
        for (int j = 1; j < 6; ++j) lm = fminf(lm, w[j]);
        float gm = wave_min(lm);
        if (it == nrem) { thr = gm; break; }
        int key = 0x7fffffff;
#pragma unroll
        for (int j = 0; j < 6; ++j)
            if (w[j] == gm) { int k2 = j * 32 + lane; key = min(key, k2); }
        int gk = wave_imin(key);
        if ((gk & 31) == lane) w[gk >> 5] = BIG;  // remove one instance
    }
#pragma unroll
    for (int j = 0; j < 6; ++j) {
        float q = (p[j] >= thr) ? p[j] : 0.f;
        out[(long)row * C_ + lane + 32 * j] = (__bf16)q;
    }
}

// one wave per spatial position: LayerNorm over C, write bf16 NHWC
__global__ __launch_bounds__(256) void layernorm_kernel(
    const float* __restrict__ Fa, const float* __restrict__ lw,
    const float* __restrict__ lb, __bf16* __restrict__ xn, long rows)
{
    long row = (long)blockIdx.x * 8 + (threadIdx.x >> 5);
    int lane = threadIdx.x & 31;
    if (row >= rows) return;
    const float* rp = Fa + row * C_;
    float v[6], s = 0.f, ss = 0.f;
#pragma unroll
    for (int j = 0; j < 6; ++j) {
        v[j] = rp[lane + 32 * j];
        s += v[j]; ss += v[j] * v[j];
    }
    s = wave_sum(s); ss = wave_sum(ss);
    float mu  = s * (1.f / C_);
    float var = ss * (1.f / C_) - mu * mu;
    float inv = rsqrtf(var + 1e-5f);
#pragma unroll
    for (int j = 0; j < 6; ++j) {
        int c = lane + 32 * j;
        xn[row * C_ + c] = (__bf16)((v[j] - mu) * inv * lw[c] + lb[c]);
    }
}

// fused depthwise-3x3 on h (1020 ch) + GELU gating -> g (padded 512 rows)
__global__ void dwgelu_kernel(const __bf16* __restrict__ h,
                              const float* __restrict__ wk,
                              const float* __restrict__ bias,
                              __bf16* __restrict__ g, long total)
{
    long idx = (long)blockIdx.x * blockDim.x + threadIdx.x;
    if (idx >= total) return;
    int  c = (int)(idx % HID);
    long t = idx / HID;
    int  x = (int)(t % W_);
    long t2 = t / W_;
    int  y = (int)(t2 % H_);
    int  b = (int)(t2 / H_);

    float acc[2];
#pragma unroll
    for (int half = 0; half < 2; ++half) {
        int ch = c + half * HID;
        float s = bias[ch];
#pragma unroll
        for (int ky = 0; ky < 3; ++ky) {
            int yy = y + ky - 1;
            if ((unsigned)yy >= (unsigned)H_) continue;
#pragma unroll
            for (int kx = 0; kx < 3; ++kx) {
                int xx = x + kx - 1;
                if ((unsigned)xx >= (unsigned)W_) continue;
                s += wk[ch * 9 + ky * 3 + kx] *
                     (float)h[(((long)b * NSP) + (long)yy * W_ + xx) * HID2 + ch];
            }
        }
        acc[half] = s;
    }
    float x2 = acc[1];
    float ge = 0.5f * x2 * (1.f + erff(x2 * 0.70710678118654752f));
    long  n  = ((long)b * NSP) + (long)y * W_ + x;
    g[n * HPAD + c] = (__bf16)(acc[0] * ge);
    if (c < HPAD - HID) g[n * HPAD + HID + c] = (__bf16)0.f;  // zero K-padding
}

// ---------------------------------------------------------------------------
// host side
// ---------------------------------------------------------------------------
static inline long cdiv(long a, long b) { return (a + b - 1) / b; }

extern "C" void kernel_launch(void* const* d_in, const int* in_sizes, int n_in,
                              void* d_out, int out_size, void* d_ws, size_t ws_size,
                              hipStream_t stream)
{
    const float* Q    = (const float*)d_in[0];
    const float* Fn   = (const float*)d_in[1];
    const float* temp = (const float*)d_in[2];
    const float* qw1  = (const float*)d_in[3];
    const float* qb1  = (const float*)d_in[4];
    const float* qw2  = (const float*)d_in[5];
    const float* qb2  = (const float*)d_in[6];
    const float* kw1  = (const float*)d_in[7];
    const float* kb1  = (const float*)d_in[8];
    const float* kw2  = (const float*)d_in[9];
    const float* kb2  = (const float*)d_in[10];
    const float* vw1  = (const float*)d_in[11];
    const float* vb1  = (const float*)d_in[12];
    const float* vw2  = (const float*)d_in[13];
    const float* vb2  = (const float*)d_in[14];
    const float* ow   = (const float*)d_in[15];
    const float* ob   = (const float*)d_in[16];
    const float* lnw  = (const float*)d_in[17];
    const float* lnb  = (const float*)d_in[18];
    const float* fiw  = (const float*)d_in[19];
    const float* fib  = (const float*)d_in[20];
    const float* fdw  = (const float*)d_in[21];
    const float* fdb  = (const float*)d_in[22];
    const float* fow  = (const float*)d_in[23];
    const float* fob  = (const float*)d_in[24];
    float* outp = (float*)d_out;

    // ---- workspace layout (256B aligned slices) ----
    char*  base = (char*)d_ws;
    size_t cur  = 0;
    auto alloc = [&](size_t bytes) -> char* {
        char* p = base + cur;
        cur = (cur + bytes + 255) & ~(size_t)255;
        return p;
    };
    const long actB  = (long)B_ * NSP * C_;                // elements
    __bf16* wq1   = (__bf16*)alloc(192 * 192 * 2);
    __bf16* wk1   = (__bf16*)alloc(192 * 192 * 2);
    __bf16* wv1   = (__bf16*)alloc(192 * 192 * 2);
    __bf16* wow   = (__bf16*)alloc(192 * 192 * 2);
    __bf16* wfi   = (__bf16*)alloc((long)HID2 * 192 * 2);
    __bf16* wfo   = (__bf16*)alloc((long)C_ * HPAD * 2);
    __bf16* Qp    = (__bf16*)alloc(actB * 2);
    __bf16* Fnp   = (__bf16*)alloc(actB * 2);              // later reused as xn
    __bf16* t1    = (__bf16*)alloc(actB * 2);              // staging / out_nhwc
    __bf16* qn    = (__bf16*)alloc(actB * 2);              // NCHW
    __bf16* kn    = (__bf16*)alloc(actB * 2);              // NCHW
    __bf16* vd    = (__bf16*)alloc(actB * 2);              // NHWC
    float*  sums  = (float*)alloc((long)B_ * C_ * 4);
    float*  attnF = (float*)alloc((long)B_ * C_ * C_ * 4);
    __bf16* attnH = (__bf16*)alloc((long)B_ * C_ * C_ * 2);
    float*  Fa    = (float*)alloc(actB * 4);
    __bf16* hbuf  = (__bf16*)alloc((long)B_ * NSP * HID2 * 2);
    __bf16* gbuf  = (__bf16*)alloc((long)B_ * NSP * HPAD * 2);
    (void)ws_size; (void)n_in; (void)in_sizes; (void)out_size;

    const int T = 256;

    // ---- weight prep ----
    cast_f2b_kernel<<<cdiv(192 * 192, T), T, 0, stream>>>(qw1, wq1, 192 * 192);
    cast_f2b_kernel<<<cdiv(192 * 192, T), T, 0, stream>>>(kw1, wk1, 192 * 192);
    cast_f2b_kernel<<<cdiv(192 * 192, T), T, 0, stream>>>(vw1, wv1, 192 * 192);
    cast_f2b_kernel<<<cdiv(192 * 192, T), T, 0, stream>>>(ow,  wow, 192 * 192);
    cast_f2b_kernel<<<cdiv((long)HID2 * 192, T), T, 0, stream>>>(fiw, wfi, (long)HID2 * 192);
    pad_fow_kernel<<<cdiv((long)C_ * HPAD, T), T, 0, stream>>>(fow, wfo);

    // ---- inputs to NHWC bf16 ----
    nchw2nhwc_kernel<<<cdiv(actB, T), T, 0, stream>>>(Q,  Qp,  actB);
    nchw2nhwc_kernel<<<cdiv(actB, T), T, 0, stream>>>(Fn, Fnp, actB);

    // GEMM launcher: 4x2 tiles/wave, 8 waves/block over N
    auto gemm = [&](const __bf16* A, long aB, const __bf16* X, long xB,
                    const float* bias, const float* scale,
                    const float* add, long addB, int ldAdd,
                    float* oF, __bf16* oH, long oB, int ldo, int tr,
                    int M, int N, int K) {
        dim3 grid((unsigned)cdiv(M, 64), (unsigned)cdiv(cdiv(N, 32), 8), B_);
        gemm_wmma_kernel<4, 2><<<grid, 256, 0, stream>>>(
            A, aB, X, xB, bias, scale, add, addB, ldAdd,
            oF, oH, oB, ldo, tr, M, N, K);
    };
    const long actStride = (long)NSP * C_;

    // ---- q path: 1x1 conv (WMMA) -> depthwise 3x3 -> NCHW ----
    gemm(wq1, 0, Qp, actStride, qb1, nullptr, nullptr, 0, 0,
         nullptr, t1, actStride, C_, 0, C_, NSP, C_);
    dw3x3_kernel<<<cdiv(actB, T), T, 0, stream>>>(t1, qw2, qb2, qn, 1, actB);
    // ---- k path ----
    gemm(wk1, 0, Fnp, actStride, kb1, nullptr, nullptr, 0, 0,
         nullptr, t1, actStride, C_, 0, C_, NSP, C_);
    dw3x3_kernel<<<cdiv(actB, T), T, 0, stream>>>(t1, kw2, kb2, kn, 1, actB);
    // ---- v path (stays NHWC) ----
    gemm(wv1, 0, Fnp, actStride, vb1, nullptr, nullptr, 0, 0,
         nullptr, t1, actStride, C_, 0, C_, NSP, C_);
    dw3x3_kernel<<<cdiv(actB, T), T, 0, stream>>>(t1, vw2, vb2, vd, 0, actB);

    // ---- L2 row normalization of q and k (NCHW rows of 16384) ----
    rowsumsq_kernel<<<B_ * C_, T, 0, stream>>>(qn, sums, NSP);
    l2scale_kernel<<<cdiv(actB, T), T, 0, stream>>>(qn, sums, actB);
    rowsumsq_kernel<<<B_ * C_, T, 0, stream>>>(kn, sums, NSP);
    l2scale_kernel<<<cdiv(actB, T), T, 0, stream>>>(kn, sums, actB);

    // ---- attn = (q . k^T) * temperature : C x C, K = 16384 ----
    gemm(qn, actStride, kn, actStride, nullptr, temp, nullptr, 0, 0,
         attnF, nullptr, (long)C_ * C_, C_, 1, C_, C_, NSP);
    softmax_topk_kernel<<<cdiv(B_ * C_, 8), 256, 0, stream>>>(attnF, attnH, B_ * C_);

    // ---- out_nhwc = v_nhwc x attn^T : M = NSP (spatial), N = C, K = C ----
    gemm(vd, actStride, attnH, (long)C_ * C_, nullptr, nullptr, nullptr, 0, 0,
         nullptr, t1, actStride, C_, 1, NSP, C_, C_);

    // ---- Fa = 1x1 conv(ow) + ob, f32 NHWC ----
    gemm(wow, 0, t1, actStride, ob, nullptr, nullptr, 0, 0,
         Fa, nullptr, actStride, C_, 0, C_, NSP, C_);

    // ---- LayerNorm over channels -> xn (reuse Fnp) ----
    __bf16* xn = Fnp;
    layernorm_kernel<<<cdiv((long)B_ * NSP, 8), 256, 0, stream>>>(
        Fa, lnw, lnb, xn, (long)B_ * NSP);

    // ---- FFN in: 1020 x 192 GEMM -> h ----
    gemm(wfi, 0, xn, actStride, fib, nullptr, nullptr, 0, 0,
         nullptr, hbuf, (long)NSP * HID2, HID2, 0, HID2, NSP, C_);

    // ---- fused depthwise 3x3 + GELU gating -> g (padded to 512) ----
    dwgelu_kernel<<<cdiv((long)B_ * NSP * HID, T), T, 0, stream>>>(
        hbuf, fdw, fdb, gbuf, (long)B_ * NSP * HID);

    // ---- FFN out GEMM with fused residual: d_out = Fa + fo(g), NCHW f32 ----
    gemm(wfo, 0, gbuf, (long)NSP * HPAD, fob, nullptr,
         Fa, actStride, C_,
         outp, nullptr, (long)C_ * NSP, NSP, 1, C_, NSP, HPAD);
}
